// AdvancedSpikeEncoder_33071248179951
// MI455X (gfx1250) — matrix operand order repaired and checked by reading, not compiled, not run on hardware
//
#include <hip/hip_runtime.h>
#include <math.h>

// ---- shapes (fixed by the reference) ----
#define Bb 4
#define Tt 16
#define Ss 256
#define Dd 512
#define Nn 8
#define Mdim (Bb * Ss)      // 1024
#define Kdim Dd             // 512
#define Edim (Dd * Nn)      // 4096

typedef __attribute__((ext_vector_type(16))) __bf16 v16bf;
typedef __attribute__((ext_vector_type(8)))  __bf16 v8bf;
typedef __attribute__((ext_vector_type(8)))  float  v8f;

// ---------------- stage 1: f32 -> bf16 conversion ----------------
__global__ __launch_bounds__(256) void cvt_bf16_kernel(const float* __restrict__ src,
                                                       __bf16* __restrict__ dst, int n) {
    int i = blockIdx.x * blockDim.x + threadIdx.x;
    if (i < n) dst[i] = (__bf16)src[i];
}

// ---------------- stage 2: softmax of the 4 encoder weights ----------------
__global__ void softmax4_kernel(const float* __restrict__ ew, float* __restrict__ w4) {
    if (threadIdx.x == 0 && blockIdx.x == 0) {
        float m = fmaxf(fmaxf(ew[0], ew[1]), fmaxf(ew[2], ew[3]));
        float e0 = expf(ew[0] - m), e1 = expf(ew[1] - m);
        float e2 = expf(ew[2] - m), e3 = expf(ew[3] - m);
        float inv = 1.0f / (e0 + e1 + e2 + e3);
        w4[0] = e0 * inv; w4[1] = e1 * inv; w4[2] = e2 * inv; w4[3] = e3 * inv;
    }
}

// ---------------- stage 3: WMMA GEMM  pop_rates = sigmoid(x @ pop_w^T + b) ----------------
// One wave per 16x16 tile of C[M=1024, E=4096]. A = x (bf16, row-major MxK),
// B(k,n) = pop_w[e0+n, k] (pop_w is row-major ExK, so B columns are contiguous rows of W).
__global__ __launch_bounds__(256) void pop_gemm_wmma_kernel(
    const __bf16* __restrict__ Abf, const __bf16* __restrict__ Wbf,
    const float* __restrict__ bias, float* __restrict__ pop_rates)
{
    const int wavesPerBlock = 256 / 32;
    int gwave = blockIdx.x * wavesPerBlock + (threadIdx.x >> 5);
    int lane  = threadIdx.x & 31;

    const int tilesE = Edim / 16;            // 256
    int tm = gwave / tilesE;                 // 0..63
    int te = gwave % tilesE;                 // 0..255
    int m0 = tm * 16;
    int e0 = te * 16;

    int half = lane >> 4;                    // 0: lanes 0-15, 1: lanes 16-31
    int r    = lane & 15;

    // A fragment (16x32 bf16): lane holds row M=r.
    //   elems 0..15 -> K = k0 + half*16 + i
    const __bf16* Arow = Abf + (size_t)(m0 + r) * Kdim;
    // B fragment (32x16 bf16): lane holds col n=r, K = k0 + half*16 + i (i=0..15),
    // with B(k, n) = W[e0+n][k] contiguous along k.
    const __bf16* Brow = Wbf + (size_t)(e0 + r) * Kdim + half * 16;

    v8f acc = {};
#pragma unroll
    for (int k0 = 0; k0 < Kdim; k0 += 32) {
        v16bf afrag = *(const v16bf*)(Arow + k0 + half * 16);
        v16bf bfrag = *(const v16bf*)(Brow + k0);
        acc = __builtin_amdgcn_wmma_f32_16x16x32_bf16(
            /*neg_a=*/false, afrag, /*neg_b=*/false, bfrag,
            /*c_mod=*/(short)0, acc, /*reuse_a=*/false, /*reuse_b=*/false);
    }

    // C/D layout: lane col N = lane&15; VGPR v -> row M = (lane>>4)*8 + v.
    int col = e0 + r;
    float bv = bias[col];
    int rowbase = m0 + half * 8;
#pragma unroll
    for (int v = 0; v < 8; ++v) {
        float val = acc[v] + bv;
        val = 1.0f / (1.0f + expf(-val));      // sigmoid -> pop firing rate
        pop_rates[(size_t)(rowbase + v) * Edim + col] = val;
    }
}

// ---------------- stage 4: fused elementwise encoders + weighted sum ----------------
__global__ __launch_bounds__(256) void fuse_kernel(
    const float* __restrict__ x, const float* __restrict__ freq,
    const float* __restrict__ noise, const float* __restrict__ rand_rate,
    const float* __restrict__ rand_pop, const float* __restrict__ pop_rates,
    const float* __restrict__ w4, float* __restrict__ out)
{
    const long total = (long)Bb * Tt * Ss * Dd;
    long idx = (long)blockIdx.x * blockDim.x + threadIdx.x;
    if (idx >= total) return;

    int d = (int)(idx % Dd);
    long tmp = idx / Dd;
    int s = (int)(tmp % Ss); tmp /= Ss;
    int t = (int)(tmp % Tt);
    int b = (int)(tmp / Tt);

    long bsd = ((long)(b * Ss + s)) * Dd + d;
    float xv  = x[bsd];
    float sig = 1.0f / (1.0f + expf(-xv));

    // RateEncoder
    float rate = sig * 0.9f + 0.05f + noise[bsd] * 0.1f;
    rate = fminf(fmaxf(rate, 0.0f), 1.0f);
    float spike_r = (rand_rate[idx] < rate) ? 1.0f : 0.0f;

    // TemporalEncoder (truncating cast, matches .astype(int32))
    int st = (int)(sig * (float)(Tt - 1));
    float spike_t = (t == st) ? 1.0f : 0.0f;

    // PopulationEncoder: mean over 8 Bernoulli draws
    const float4* pr = (const float4*)(pop_rates + bsd * Nn);
    const float4* rp = (const float4*)(rand_pop + idx * Nn);
    float4 p0 = pr[0], p1 = pr[1];
    float4 q0 = rp[0], q1 = rp[1];
    float cnt = (q0.x < p0.x ? 1.0f : 0.0f) + (q0.y < p0.y ? 1.0f : 0.0f)
              + (q0.z < p0.z ? 1.0f : 0.0f) + (q0.w < p0.w ? 1.0f : 0.0f)
              + (q1.x < p1.x ? 1.0f : 0.0f) + (q1.y < p1.y ? 1.0f : 0.0f)
              + (q1.z < p1.z ? 1.0f : 0.0f) + (q1.w < p1.w ? 1.0f : 0.0f);
    float spike_p = cnt * 0.125f;

    // PhaseEncoder: t in linspace(0, 2pi, 16)
    const float TWO_PI = 6.283185307179586f;
    float tv = TWO_PI * (float)t / (float)(Tt - 1);
    float wave = sinf(freq[d] * tv + sig * TWO_PI);
    float spike_ph = (wave > 0.5f) ? 1.0f : 0.0f;

    out[idx] = w4[0] * spike_r + w4[1] * spike_t + w4[2] * spike_p + w4[3] * spike_ph;
}

// ---------------- launch ----------------
extern "C" void kernel_launch(void* const* d_in, const int* in_sizes, int n_in,
                              void* d_out, int out_size, void* d_ws, size_t ws_size,
                              hipStream_t stream) {
    const float* x         = (const float*)d_in[0];   // [B,S,D]
    const float* freq      = (const float*)d_in[1];   // [D]
    const float* pop_w     = (const float*)d_in[2];   // [D*N, D]
    const float* pop_b     = (const float*)d_in[3];   // [D*N]
    const float* encw      = (const float*)d_in[4];   // [4]
    const float* noise     = (const float*)d_in[5];   // [B,S,D]
    const float* rand_rate = (const float*)d_in[6];   // [B,T,S,D]
    const float* rand_pop  = (const float*)d_in[7];   // [B,T,S,D,N]
    float* out = (float*)d_out;

    // workspace layout (all 256B aligned):
    //   Abf  : bf16 x,      1024*512*2  = 1 MiB
    //   Wbf  : bf16 pop_w,  4096*512*2  = 4 MiB
    //   prat : f32 rates,   1024*4096*4 = 16 MiB
    //   w4   : f32 softmax weights, 16 B
    char* ws = (char*)d_ws;
    __bf16* Abf  = (__bf16*)ws;
    __bf16* Wbf  = (__bf16*)(ws + (1u << 20));
    float*  prat = (float*)(ws + (1u << 20) + (4u << 20));
    float*  w4   = (float*)(ws + (1u << 20) + (4u << 20) + (16u << 20));

    const int nX = Mdim * Kdim;       // 524288
    const int nW = Edim * Kdim;       // 2097152
    cvt_bf16_kernel<<<(nX + 255) / 256, 256, 0, stream>>>(x, Abf, nX);
    cvt_bf16_kernel<<<(nW + 255) / 256, 256, 0, stream>>>(pop_w, Wbf, nW);
    softmax4_kernel<<<1, 32, 0, stream>>>(encw, w4);

    // 64 * 256 = 16384 wave-tiles, 8 waves (256 threads) per block
    int tiles = (Mdim / 16) * (Edim / 16);
    pop_gemm_wmma_kernel<<<tiles / 8, 256, 0, stream>>>(Abf, Wbf, pop_b, prat);

    long total = (long)Bb * Tt * Ss * Dd;  // 8388608
    fuse_kernel<<<(int)((total + 255) / 256), 256, 0, stream>>>(
        x, freq, noise, rand_rate, rand_pop, prat, w4, out);
}
